// TransformerBlock_50749333569900
// MI455X (gfx1250) — matrix operand order repaired and checked by reading, not compiled, not run on hardware
//
#include <hip/hip_runtime.h>
#include <hip/hip_bf16.h>

// ---------------------------------------------------------------------------
// Transformer block for MI455X (gfx1250, wave32, WMMA + async-to-LDS DMA).
// All GEMM-like einsums run on V_WMMA_F32_16X16X32_F16; GEMM LDS staging uses
// GLOBAL_LOAD_ASYNC_TO_LDS_B128 with double buffering (ASYNCcnt-tracked).
// ---------------------------------------------------------------------------

typedef _Float16 half_t;
typedef __attribute__((ext_vector_type(8)))  _Float16 v8h;
typedef __attribute__((ext_vector_type(16))) _Float16 v16h;
typedef __attribute__((ext_vector_type(8)))  float    v8f;
typedef __attribute__((ext_vector_type(4)))  float    v4f;

#define D_MODEL 2048
#define N_HEADS 16
#define D_K     128
#define D_FF    5504
#define BATCH   2
#define SEQ     2048
#define M_ROWS  (BATCH * SEQ)          // 4096

static __device__ __forceinline__ v8f zero8() {
  v8f r;
#pragma unroll
  for (int i = 0; i < 8; ++i) r[i] = 0.0f;
  return r;
}

static __device__ __forceinline__ v16h join16(v8h lo, v8h hi) {
  v16h r;
#pragma unroll
  for (int i = 0; i < 8; ++i) { r[i] = lo[i]; r[i + 8] = hi[i]; }
  return r;
}

// Async DMA: copy 16B from global to LDS, tracked by ASYNCcnt.
// lds_off: low 32 bits of generic LDS pointer == wave-relative LDS address.
static __device__ __forceinline__ void async_copy_b128(uint32_t lds_off,
                                                       const void* gptr) {
  asm volatile("global_load_async_to_lds_b128 %0, %1, off"
               :: "v"(lds_off), "v"((unsigned long long)(uintptr_t)gptr)
               : "memory");
}

static __device__ __forceinline__ void wait_async0() {
  asm volatile("s_wait_asynccnt 0x0" ::: "memory");
}

// ---------------------------------------------------------------------------
// f32 -> f16 weight conversion (one pass per weight, 4 elems/thread)
// ---------------------------------------------------------------------------
__global__ __launch_bounds__(256) void cvt_f16_kernel(
    const float* __restrict__ src, half_t* __restrict__ dst, size_t n4) {
  const size_t i = ((size_t)blockIdx.x * 256 + threadIdx.x);
  if (i >= n4) return;
  v4f a = *(const v4f*)(src + i * 4);
  half_t* d = dst + i * 4;
#pragma unroll
  for (int j = 0; j < 4; ++j) d[j] = (half_t)a[j];
}

// ---------------------------------------------------------------------------
// RMSNorm: per row of D elements, f32 in, f16 out (feeds WMMA A operand)
// ---------------------------------------------------------------------------
__global__ __launch_bounds__(256) void rmsnorm_f16_kernel(
    const float* __restrict__ x, const float* __restrict__ g,
    half_t* __restrict__ out, int D) {
  __shared__ float red[256];
  const int row = blockIdx.x;
  const float* xr = x + (size_t)row * D;
  float s = 0.0f;
  for (int j = threadIdx.x; j < D; j += 256) { float v = xr[j]; s += v * v; }
  red[threadIdx.x] = s;
  __syncthreads();
  for (int off = 128; off > 0; off >>= 1) {
    if (threadIdx.x < off) red[threadIdx.x] += red[threadIdx.x + off];
    __syncthreads();
  }
  const float inv = rsqrtf(red[0] / (float)D + 1e-5f);
  half_t* orow = out + (size_t)row * D;
  for (int j = threadIdx.x; j < D; j += 256)
    orow[j] = (half_t)(xr[j] * inv * g[j]);
}

// ---------------------------------------------------------------------------
// WMMA GEMM: C[M,N] = A[M,K](f16) * B[N,K](f16)^T
// Block tile 128x128, K-step 32, 8 waves (wave w owns rows [w*16, w*16+16)).
// Global->LDS staging via async DMA, double buffered: DMA of tile k+1
// overlaps the 8 WMMAs of tile k.
// Optional: resid (f32, added), Cf (f32 out), Ch (f16 out).
// ---------------------------------------------------------------------------
#define LDSS 40  // half_t stride (80B: 16B-aligned fragments, staggers banks)

__global__ __launch_bounds__(256) void gemm_wmma_kernel(
    const half_t* __restrict__ A, const half_t* __restrict__ Bh,
    const float* __restrict__ resid, float* __restrict__ Cf,
    half_t* __restrict__ Ch, int M, int N, int K) {
  __shared__ half_t As[2][128 * LDSS];
  __shared__ half_t Bs[2][128 * LDSS];

  const int tid  = threadIdx.x;
  const int lane = tid & 31;
  const int wave = tid >> 5;
  const int bm   = blockIdx.y * 128;
  const int bn   = blockIdx.x * 128;

  v8f acc[8];
#pragma unroll
  for (int i = 0; i < 8; ++i) acc[i] = zero8();

  // DMA mapping: thread covers rows {r64, r64+64}, 16B chunk chk of 64B row
  const int r64 = tid >> 2;        // 0..63
  const int chk = (tid & 3) * 8;   // half_t offset of 16B chunk

  auto stage = [&](int buf, int k0) {
#pragma unroll
    for (int h = 0; h < 2; ++h) {
      const int row = h * 64 + r64;
      async_copy_b128((uint32_t)(uintptr_t)&As[buf][row * LDSS + chk],
                      A + (size_t)(bm + row) * K + k0 + chk);
    }
#pragma unroll
    for (int h = 0; h < 2; ++h) {
      const int row = h * 64 + r64;
      async_copy_b128((uint32_t)(uintptr_t)&Bs[buf][row * LDSS + chk],
                      Bh + (size_t)(bn + row) * K + k0 + chk);
    }
  };

  const int kb  = (lane < 16) ? 0 : 16;   // A fragment K base (contiguous layout)
  const int kb2 = (lane < 16) ? 0 : 16;   // B fragment K base
  const int l15 = lane & 15;
  const int am  = wave * 16 + l15;

  // prologue: fill buffer 0
  stage(0, 0);
  wait_async0();
  __syncthreads();

  int buf = 0;
  for (int k0 = 0; k0 < K; k0 += 32) {
    if (k0 + 32 < K) stage(buf ^ 1, k0 + 32);  // DMA next tile (overlapped)

    v16h afrag = join16(*(const v8h*)&As[buf][am * LDSS + kb],
                        *(const v8h*)&As[buf][am * LDSS + kb + 8]);
#pragma unroll
    for (int nt = 0; nt < 8; ++nt) {
      const int bnr = nt * 16 + l15;
      v16h bfrag = join16(*(const v8h*)&Bs[buf][bnr * LDSS + kb2],
                          *(const v8h*)&Bs[buf][bnr * LDSS + kb2 + 8]);
      acc[nt] = __builtin_amdgcn_wmma_f32_16x16x32_f16(
          false, afrag, false, bfrag, (short)0, acc[nt], false, false);
    }
    wait_async0();     // next tile landed in LDS
    __syncthreads();   // all waves done reading buf / writing buf^1
    buf ^= 1;
  }

  // epilogue: C layout -> lanes 0-15 rows r, lanes 16-31 rows r+8
  const int rowOff = (lane & 16) ? 8 : 0;
#pragma unroll
  for (int nt = 0; nt < 8; ++nt) {
    const int col = bn + nt * 16 + l15;
#pragma unroll
    for (int r = 0; r < 8; ++r) {
      const int row = bm + wave * 16 + r + rowOff;
      float v = acc[nt][r];
      const size_t idx = (size_t)row * N + col;
      if (resid) v += resid[idx];
      if (Cf) Cf[idx] = v;
      if (Ch) Ch[idx] = (half_t)v;
    }
  }
}

// ---------------------------------------------------------------------------
// RoPE + head packing.
// q,k,v: f16 [B,S,2048]. Writes Qh,Kh f16 [B,H,S,128] and Vt f16 [B,H,128,S]
// (V transposed so the P*V WMMA B-fragment is a contiguous 32B load).
// ---------------------------------------------------------------------------
__global__ __launch_bounds__(256) void rope_pack_kernel(
    const half_t* __restrict__ q, const half_t* __restrict__ k,
    const half_t* __restrict__ v, half_t* __restrict__ Qh,
    half_t* __restrict__ Kh, half_t* __restrict__ Vt) {
  size_t i = (size_t)blockIdx.x * 256 + threadIdx.x;
  const size_t total = (size_t)BATCH * SEQ * N_HEADS * (D_K / 2);
  if (i >= total) return;
  const int p = (int)(i & 63); i >>= 6;
  const int h = (int)(i % N_HEADS); i /= N_HEADS;
  const int s = (int)(i % SEQ); i /= SEQ;
  const int b = (int)i;

  const float freq  = __expf((float)p * (-2.0f / 128.0f) * 9.210340372f);
  const float phase = (float)s * freq;
  const float cs = __cosf(phase), sn = __sinf(phase);

  const size_t src = ((size_t)b * SEQ + s) * D_MODEL + h * D_K + 2 * p;
  const float qe = (float)q[src], qo = (float)q[src + 1];
  const float ke = (float)k[src], ko = (float)k[src + 1];

  const size_t dqk = (((size_t)b * N_HEADS + h) * SEQ + s) * D_K + 2 * p;
  Qh[dqk]     = (half_t)(qe * cs - qo * sn);
  Qh[dqk + 1] = (half_t)(qe * sn + qo * cs);
  Kh[dqk]     = (half_t)(ke * cs - ko * sn);
  Kh[dqk + 1] = (half_t)(ke * sn + ko * cs);

  const size_t dv = (((size_t)b * N_HEADS + h) * D_K + 2 * p) * SEQ + s;
  Vt[dv]       = v[src];
  Vt[dv + SEQ] = v[src + 1];
}

// ---------------------------------------------------------------------------
// Flash attention: one wave per (b, h, 16-row Q tile).
// Scores via WMMA (Q as A, K-tile^T as B), online softmax through per-wave
// LDS (also transposes P from C layout to A layout), P*V via WMMA with Vt.
// ---------------------------------------------------------------------------
__global__ __launch_bounds__(32) void attn_wmma_kernel(
    const half_t* __restrict__ Qh, const half_t* __restrict__ Kh,
    const half_t* __restrict__ Vt, half_t* __restrict__ Out) {
  __shared__ float  Sbuf[16][32];
  __shared__ half_t Pbuf[16][32];
  __shared__ float  mrow[16], lrow[16], arow[16];

  const int lane = threadIdx.x;
  const int qb = blockIdx.x * 16;
  const int h  = blockIdx.y;
  const int b  = blockIdx.z;

  const size_t head = ((size_t)b * N_HEADS + h) * SEQ * D_K;
  const int kb   = (lane < 16) ? 0 : 16;
  const int kb2  = (lane < 16) ? 0 : 16;
  const int l15  = lane & 15;
  const int rowOff = (lane & 16) ? 8 : 0;
  const float scale = 0.0883883476f;  // 1/sqrt(128)

  v16h qf[4];
#pragma unroll
  for (int c = 0; c < 4; ++c) {
    const half_t* src = Qh + head + (size_t)(qb + l15) * D_K + c * 32;
    qf[c] = join16(*(const v8h*)(src + kb), *(const v8h*)(src + kb + 8));
  }

  v8f o[8];
#pragma unroll
  for (int i = 0; i < 8; ++i) o[i] = zero8();
  if (lane < 16) { mrow[lane] = -__builtin_inff(); lrow[lane] = 0.0f; }
  __syncthreads();

  for (int kt = 0; kt * 32 <= qb + 15; ++kt) {
    v8f s0 = zero8(), s1 = zero8();
#pragma unroll
    for (int c = 0; c < 4; ++c) {
      const half_t* k0 = Kh + head + (size_t)(kt * 32 + l15) * D_K + c * 32 + kb2;
      v16h bf0 = join16(*(const v8h*)(k0), *(const v8h*)(k0 + 8));
      s0 = __builtin_amdgcn_wmma_f32_16x16x32_f16(false, qf[c], false, bf0,
                                                  (short)0, s0, false, false);
      const half_t* k1 = k0 + 16 * D_K;
      v16h bf1 = join16(*(const v8h*)(k1), *(const v8h*)(k1 + 8));
      s1 = __builtin_amdgcn_wmma_f32_16x16x32_f16(false, qf[c], false, bf1,
                                                  (short)0, s1, false, false);
    }
#pragma unroll
    for (int r = 0; r < 8; ++r) {
      const int row = r + rowOff;
      const int qpos = qb + row;
      const int key0 = kt * 32 + l15;
      float v0 = s0[r] * scale; if (key0 > qpos)      v0 = -__builtin_inff();
      float v1 = s1[r] * scale; if (key0 + 16 > qpos) v1 = -__builtin_inff();
      Sbuf[row][l15]      = v0;
      Sbuf[row][l15 + 16] = v1;
    }
    __syncthreads();
    if (lane < 16) {
      const float mold = mrow[lane], lold = lrow[lane];
      float mloc = -__builtin_inff();
#pragma unroll
      for (int j = 0; j < 32; ++j) mloc = fmaxf(mloc, Sbuf[lane][j]);
      const float mnew = fmaxf(mold, mloc);
      const float alpha = __expf(mold - mnew);
      float psum = 0.0f;
#pragma unroll
      for (int j = 0; j < 32; ++j) {
        const float p = __expf(Sbuf[lane][j] - mnew);
        Pbuf[lane][j] = (half_t)p;
        psum += p;
      }
      mrow[lane] = mnew;
      lrow[lane] = lold * alpha + psum;
      arow[lane] = alpha;
    }
    __syncthreads();
#pragma unroll
    for (int r = 0; r < 8; ++r) {
      const float al = arow[r + rowOff];
#pragma unroll
      for (int nt = 0; nt < 8; ++nt) o[nt][r] *= al;
    }
    v16h pa = join16(*(const v8h*)&Pbuf[l15][kb],
                     *(const v8h*)&Pbuf[l15][kb + 8]);
#pragma unroll
    for (int nt = 0; nt < 8; ++nt) {
      const int d = nt * 16 + l15;
      const half_t* vs = Vt + head + (size_t)d * SEQ + kt * 32 + kb2;
      v16h vb = join16(*(const v8h*)(vs), *(const v8h*)(vs + 8));
      o[nt] = __builtin_amdgcn_wmma_f32_16x16x32_f16(false, pa, false, vb,
                                                     (short)0, o[nt], false, false);
    }
    __syncthreads();
  }

#pragma unroll
  for (int r = 0; r < 8; ++r) {
    const int row = r + rowOff;
    const float inv = 1.0f / lrow[row];
    const size_t base = ((size_t)b * SEQ + (qb + row)) * D_MODEL + h * D_K;
#pragma unroll
    for (int nt = 0; nt < 8; ++nt)
      Out[base + nt * 16 + l15] = (half_t)(o[nt][r] * inv);
  }
}

// ---------------------------------------------------------------------------
// SwiGLU: h = w1x * sigmoid(w1x) * w3x, in-place into w1x
// ---------------------------------------------------------------------------
__global__ __launch_bounds__(256) void swiglu_kernel(
    half_t* __restrict__ w1, const half_t* __restrict__ w3, size_t n) {
  const size_t i = (size_t)blockIdx.x * 256 + threadIdx.x;
  if (i >= n) return;
  const float a = (float)w1[i];
  const float c = (float)w3[i];
  w1[i] = (half_t)(a / (1.0f + __expf(-a)) * c);
}

// ---------------------------------------------------------------------------
// Launch. Workspace pool (~257 MB):
//   [0]        xn / xn2 f16                       16.78 MB
//   [+1*SZ]    q,k,v f16 (3x)  -> reused as w1x/h
//   [+4*SZ]    Qh,Kh,Vt f16 (3x) -> reused as w3x
//   [+7*SZ]    attn f16                           16.78 MB
//   [+8*SZ]    x_res f32                          33.55 MB
//   [+10*SZ]   f16 weights: Wq,Wk,Wv,Wo (8.39MB ea), W1,W2,W3 (22.5MB ea)
// ---------------------------------------------------------------------------
extern "C" void kernel_launch(void* const* d_in, const int* in_sizes, int n_in,
                              void* d_out, int out_size, void* d_ws, size_t ws_size,
                              hipStream_t stream) {
  (void)in_sizes; (void)n_in; (void)out_size; (void)ws_size;
  const float* x  = (const float*)d_in[0];
  const float* Wq = (const float*)d_in[1];
  const float* Wk = (const float*)d_in[2];
  const float* Wv = (const float*)d_in[3];
  const float* Wo = (const float*)d_in[4];
  const float* W1 = (const float*)d_in[5];
  const float* W2 = (const float*)d_in[6];
  const float* W3 = (const float*)d_in[7];
  const float* g1 = (const float*)d_in[8];
  const float* g2 = (const float*)d_in[9];

  char* ws = (char*)d_ws;
  const size_t SZ   = (size_t)M_ROWS * D_MODEL * sizeof(half_t);   // 16.78 MB
  const size_t SZWD = (size_t)D_MODEL * D_MODEL * sizeof(half_t);  //  8.39 MB
  const size_t SZWF = (size_t)D_FF * D_MODEL * sizeof(half_t);     // 22.54 MB

  half_t* xn   = (half_t*)(ws);
  half_t* qb   = (half_t*)(ws + SZ);
  half_t* kbuf = (half_t*)(ws + 2 * SZ);
  half_t* vbuf = (half_t*)(ws + 3 * SZ);
  half_t* Qh   = (half_t*)(ws + 4 * SZ);
  half_t* Kh   = (half_t*)(ws + 5 * SZ);
  half_t* Vt   = (half_t*)(ws + 6 * SZ);
  half_t* attn = (half_t*)(ws + 7 * SZ);
  float*  xres = (float*)(ws + 8 * SZ);            // 2*SZ bytes
  char*   wp   = ws + 10 * SZ;
  half_t* Wqh  = (half_t*)(wp);
  half_t* Wkh  = (half_t*)(wp + SZWD);
  half_t* Wvh  = (half_t*)(wp + 2 * SZWD);
  half_t* Woh  = (half_t*)(wp + 3 * SZWD);
  half_t* W1h  = (half_t*)(wp + 4 * SZWD);
  half_t* W2h  = (half_t*)(wp + 4 * SZWD + SZWF);
  half_t* W3h  = (half_t*)(wp + 4 * SZWD + 2 * SZWF);
  half_t* w1x  = qb;   // reuse qkv region (50.3MB >= 45.1MB)
  half_t* w3x  = Qh;   // reuse heads region

  const dim3 blk256(256);
  const dim3 gGemmD(D_MODEL / 128, M_ROWS / 128);  // (16,32)
  const dim3 gGemmF(D_FF / 128,   M_ROWS / 128);   // (43,32)

  // 0) one-time f32 -> f16 weight conversion
  const size_t nD4 = (size_t)D_MODEL * D_MODEL / 4;
  const size_t nF4 = (size_t)D_FF * D_MODEL / 4;
  cvt_f16_kernel<<<(unsigned)(nD4 / 256), blk256, 0, stream>>>(Wq, Wqh, nD4);
  cvt_f16_kernel<<<(unsigned)(nD4 / 256), blk256, 0, stream>>>(Wk, Wkh, nD4);
  cvt_f16_kernel<<<(unsigned)(nD4 / 256), blk256, 0, stream>>>(Wv, Wvh, nD4);
  cvt_f16_kernel<<<(unsigned)(nD4 / 256), blk256, 0, stream>>>(Wo, Woh, nD4);
  cvt_f16_kernel<<<(unsigned)(nF4 / 256), blk256, 0, stream>>>(W1, W1h, nF4);
  cvt_f16_kernel<<<(unsigned)(nF4 / 256), blk256, 0, stream>>>(W2, W2h, nF4);
  cvt_f16_kernel<<<(unsigned)(nF4 / 256), blk256, 0, stream>>>(W3, W3h, nF4);

  // 1) xn = rmsnorm(x, g1)
  rmsnorm_f16_kernel<<<M_ROWS, blk256, 0, stream>>>(x, g1, xn, D_MODEL);
  // 2-4) Q,K,V projections
  gemm_wmma_kernel<<<gGemmD, blk256, 0, stream>>>(xn, Wqh, nullptr, nullptr, qb,
                                                  M_ROWS, D_MODEL, D_MODEL);
  gemm_wmma_kernel<<<gGemmD, blk256, 0, stream>>>(xn, Wkh, nullptr, nullptr, kbuf,
                                                  M_ROWS, D_MODEL, D_MODEL);
  gemm_wmma_kernel<<<gGemmD, blk256, 0, stream>>>(xn, Wvh, nullptr, nullptr, vbuf,
                                                  M_ROWS, D_MODEL, D_MODEL);
  // 5) RoPE + head packing (V transposed)
  {
    const size_t total = (size_t)BATCH * SEQ * N_HEADS * (D_K / 2);
    rope_pack_kernel<<<(unsigned)(total / 256), blk256, 0, stream>>>(
        qb, kbuf, vbuf, Qh, Kh, Vt);
  }
  // 6) causal flash attention
  attn_wmma_kernel<<<dim3(SEQ / 16, N_HEADS, BATCH), dim3(32), 0, stream>>>(
      Qh, Kh, Vt, attn);
  // 7) x_res = x + attn @ Wo^T
  gemm_wmma_kernel<<<gGemmD, blk256, 0, stream>>>(attn, Woh, x, xres, nullptr,
                                                  M_ROWS, D_MODEL, D_MODEL);
  // 8) xn2 = rmsnorm(x_res, g2)
  rmsnorm_f16_kernel<<<M_ROWS, blk256, 0, stream>>>(xres, g2, xn, D_MODEL);
  // 9-10) W1, W3 projections
  gemm_wmma_kernel<<<gGemmF, blk256, 0, stream>>>(xn, W1h, nullptr, nullptr, w1x,
                                                  M_ROWS, D_FF, D_MODEL);
  gemm_wmma_kernel<<<gGemmF, blk256, 0, stream>>>(xn, W3h, nullptr, nullptr, w3x,
                                                  M_ROWS, D_FF, D_MODEL);
  // 11) h = silu(w1x) * w3x  (in place into w1x)
  {
    const size_t n = (size_t)M_ROWS * D_FF;
    swiglu_kernel<<<(unsigned)((n + 255) / 256), blk256, 0, stream>>>(w1x, w3x, n);
  }
  // 12) out = x_res + h @ W2^T
  gemm_wmma_kernel<<<gGemmD, blk256, 0, stream>>>(w1x, W2h, xres, (float*)d_out,
                                                  nullptr, M_ROWS, D_MODEL, D_FF);
}
